// GCN_80238579024339
// MI455X (gfx1250) — compile-verified, matrix operand order checked by reading, not run on hardware
//
#include <hip/hip_runtime.h>
#include <hip/hip_bf16.h>
#include <math.h>

typedef __attribute__((ext_vector_type(16))) _Float16 v16h;
typedef __attribute__((ext_vector_type(8)))  _Float16 v8h;
typedef __attribute__((ext_vector_type(8)))  float    v8f;

#define GG    256
#define NN    512
#define EE    4096
#define FIN   64
#define FOUT  64
#define CCLS  16
#define KFLAT (NN * FOUT)   // 32768

// ---------------------------------------------------------------------------
// Kernel 1: xw[g,n,o] = sum_k x[g,n,k] * W[o,k]   (f32 in, f16 out, WMMA f16)
// 1024 blocks x 256 threads; each wave -> one 16-row tile, all 64 cols.
// ---------------------------------------------------------------------------
__global__ void __launch_bounds__(256)
gcn_xw_gemm(const float* __restrict__ x, const float* __restrict__ w,
            _Float16* __restrict__ xw) {
  const int tid  = threadIdx.x;
  const int lane = tid & 31;
  const int wv   = tid >> 5;
  const int tile = blockIdx.x * 8 + wv;       // 8192 tiles = 256 graphs * 32
  const int g    = tile >> 5;
  const int m0   = (tile & 31) << 4;
  const int mr   = lane & 15;
  const int kh   = lane >> 4;                 // 0 or 1

  // A fragments (16x32 f16): halves 0..7 -> K = kc*32 + kh*8 + j
  //                          halves 8..15 -> K = kc*32 + 16 + kh*8 + j
  const float* xrow = x + ((size_t)g * NN + (size_t)(m0 + mr)) * FIN;
  v16h a[2];
#pragma unroll
  for (int kc = 0; kc < 2; ++kc) {
    const float* p0 = xrow + kc * 32 + kh * 8;
    const float* p1 = p0 + 16;
#pragma unroll
    for (int j = 0; j < 8; ++j) {
      a[kc][j]     = (_Float16)p0[j];
      a[kc][j + 8] = (_Float16)p1[j];
    }
  }

  v8f acc[4] = {};
#pragma unroll
  for (int nt = 0; nt < 4; ++nt) {
    // B[k, o] = W[o, k]; this lane supplies column o = nt*16 + mr
    const float* wrow = w + (size_t)(nt * 16 + mr) * FIN;
#pragma unroll
    for (int kc = 0; kc < 2; ++kc) {
      v16h b;
      const float* q0 = wrow + kc * 32 + kh * 8;
      const float* q1 = q0 + 16;
#pragma unroll
      for (int j = 0; j < 8; ++j) {
        b[j]     = (_Float16)q0[j];
        b[j + 8] = (_Float16)q1[j];
      }
      acc[nt] = __builtin_amdgcn_wmma_f32_16x16x32_f16(
          false, a[kc], false, b, (short)0, acc[nt], false, false);
    }
  }

  // D layout: VGPR r, lanes 0-15 -> M=r, lanes 16-31 -> M=8+r; N = lane&15
  const int rb = kh * 8;
#pragma unroll
  for (int nt = 0; nt < 4; ++nt) {
#pragma unroll
    for (int r = 0; r < 8; ++r) {
      xw[((size_t)g * NN + (size_t)(m0 + rb + r)) * FOUT + nt * 16 + mr] =
          (_Float16)acc[nt][r];
    }
  }
}

// ---------------------------------------------------------------------------
// Kernel 2: GCN aggregation (symmetric norm, self-loops) + bias + ReLU.
// One block per graph; h accumulator lives in LDS (512*64 f32 = 128 KB).
// ---------------------------------------------------------------------------
__global__ void __launch_bounds__(256)
gcn_aggregate(const _Float16* __restrict__ xw, const int* __restrict__ ei,
              const float* __restrict__ bias, _Float16* __restrict__ hout) {
  __shared__ float    hacc[NN * FOUT];   // 128 KB
  __shared__ float    dinv[NN];
  __shared__ unsigned deg[NN];

  const int g   = blockIdx.x;
  const int tid = threadIdx.x;
  const int* rowp = ei + (size_t)g * 2 * EE;
  const int* colp = rowp + EE;

  for (int i = tid; i < NN; i += 256) deg[i] = 1u;          // self-loop
  for (int i = tid; i < NN * FOUT; i += 256) hacc[i] = 0.0f;
  __syncthreads();
  for (int e = tid; e < EE; e += 256) atomicAdd(&deg[colp[e]], 1u);
  __syncthreads();
  for (int i = tid; i < NN; i += 256) dinv[i] = rsqrtf((float)deg[i]);
  __syncthreads();

  const int lane = tid & 31;
  const int wv   = tid >> 5;
  const _Float16* xg = xw + (size_t)g * NN * FOUT;

  // one wave per edge; each lane covers 2 features (packed u32 load)
  for (int e = wv; e < EE + NN; e += 8) {
    int r, c;
    if (e < EE) { r = rowp[e]; c = colp[e]; }
    else        { r = e - EE;  c = r;       }
    const float nrm = dinv[r] * dinv[c];
    union { unsigned u; _Float16 f[2]; } uu;
    uu.u = *(const unsigned*)(xg + (size_t)r * FOUT + lane * 2);
    atomicAdd(&hacc[c * FOUT + lane * 2 + 0], (float)uu.f[0] * nrm);
    atomicAdd(&hacc[c * FOUT + lane * 2 + 1], (float)uu.f[1] * nrm);
  }
  __syncthreads();

  _Float16* ho = hout + (size_t)g * KFLAT;
  for (int i = tid; i < NN * FOUT; i += 256) {
    float v = hacc[i] + bias[i & (FOUT - 1)];
    ho[i] = (_Float16)fmaxf(v, 0.0f);
  }
}

// ---------------------------------------------------------------------------
// Kernel 3: logits = H[256,32768] x lin_w^T + b, then log_softmax (WMMA f16).
// 16 blocks (one per 16-graph tile); 8 waves split K; LDS reduce.
// ---------------------------------------------------------------------------
__global__ void __launch_bounds__(256)
gcn_classify(const _Float16* __restrict__ h, const float* __restrict__ lw,
             const float* __restrict__ lb, float* __restrict__ out) {
  __shared__ float red[8 * 256];
  __shared__ float lg[16 * 16];

  const int tid  = threadIdx.x;
  const int lane = tid & 31;
  const int wv   = tid >> 5;
  const int gt   = blockIdx.x;
  const int mr   = lane & 15;
  const int kh   = lane >> 4;

  const int kslice = KFLAT / 8;               // 4096 per wave
  const int kbase  = wv * kslice;

  const _Float16* arow = h + (size_t)(gt * 16 + mr) * KFLAT;
  const float*    brow = lw + (size_t)mr * KFLAT;   // class row c = mr

  v8f acc = {};
  for (int k0 = kbase; k0 < kbase + kslice; k0 += 32) {
    v16h a, b;
    const v8h lo = *(const v8h*)(arow + k0 + kh * 8);
    const v8h hi = *(const v8h*)(arow + k0 + kh * 8 + 16);
#pragma unroll
    for (int j = 0; j < 8; ++j) { a[j] = lo[j]; a[j + 8] = hi[j]; }
    const float* pb0 = brow + k0 + kh * 8;
    const float* pb1 = pb0 + 16;
#pragma unroll
    for (int j = 0; j < 8; ++j) {
      b[j]     = (_Float16)pb0[j];
      b[j + 8] = (_Float16)pb1[j];
    }
    acc = __builtin_amdgcn_wmma_f32_16x16x32_f16(
        false, a, false, b, (short)0, acc, false, false);
  }

#pragma unroll
  for (int r = 0; r < 8; ++r) red[wv * 256 + lane * 8 + r] = acc[r];
  __syncthreads();

  {
    float s = 0.0f;
#pragma unroll
    for (int w2 = 0; w2 < 8; ++w2) s += red[w2 * 256 + tid];
    const int lp = tid >> 3, r = tid & 7;
    const int M  = r + ((lp >> 4) << 3);      // graph row in tile
    const int Nc = lp & 15;                   // class
    lg[M * 16 + Nc] = s + lb[Nc];
  }
  __syncthreads();

  if (tid < 16) {
    float mx = -1e30f;
#pragma unroll
    for (int c = 0; c < 16; ++c) mx = fmaxf(mx, lg[tid * 16 + c]);
    float se = 0.0f;
#pragma unroll
    for (int c = 0; c < 16; ++c) se += expf(lg[tid * 16 + c] - mx);
    const float lse = logf(se) + mx;
    float* orow = out + (size_t)(gt * 16 + tid) * 16;
#pragma unroll
    for (int c = 0; c < 16; ++c) orow[c] = lg[tid * 16 + c] - lse;
  }
}

// ---------------------------------------------------------------------------
extern "C" void kernel_launch(void* const* d_in, const int* in_sizes, int n_in,
                              void* d_out, int out_size, void* d_ws, size_t ws_size,
                              hipStream_t stream) {
  (void)in_sizes; (void)n_in; (void)out_size; (void)ws_size;
  const float* x  = (const float*)d_in[0];   // [G, N, 64]
  const int*   ei = (const int*)  d_in[1];   // [G, 2, E]
  const float* cw = (const float*)d_in[2];   // [64, 64]
  const float* cb = (const float*)d_in[3];   // [64]
  const float* lw = (const float*)d_in[4];   // [16, 32768]
  const float* lb = (const float*)d_in[5];   // [16]
  float* out = (float*)d_out;                // [G, 16]

  _Float16* xw    = (_Float16*)d_ws;                                   // 16 MB
  _Float16* hflat = (_Float16*)((char*)d_ws + (size_t)GG * NN * FOUT * 2);

  gcn_xw_gemm  <<<1024, 256, 0, stream>>>(x, cw, xw);
  gcn_aggregate<<<GG,   256, 0, stream>>>(xw, ei, cb, hflat);
  gcn_classify <<<16,   256, 0, stream>>>(hflat, lw, lb, out);
}
